// GNN_60816736911411
// MI455X (gfx1250) — matrix operand order repaired and checked by reading, not compiled, not run on hardware
//
#include <hip/hip_runtime.h>
#include <hip/hip_bf16.h>

// ---------------------------------------------------------------------------
// GNN-CBF forward for MI455X (gfx1250, wave32, WMMA).
//
// d_in layout assumption (states, edge_index in setup_inputs() dict order,
// then params flattened as a JAX pytree: dict keys sorted, so within each
// GNN layer 'lstm' precedes 'phi'):
//   0: states[4096,8]  1: edge_index[2,524288]
//   2..15  layer1: wih0[512,256] whh0[512,128] bih0 bhh0 wih1[512,128]
//                  whh1[512,128] bih1 bhh1 | w1[136,20] b1 w2[20,20] b2
//                  w3[20,128] b3
//   16..29 layer2: same
//   30: wout[128]  31: bout[1]
// ---------------------------------------------------------------------------

#define N_AGENTS 4096
#define FDIM 128
#define PDIM 128
#define SDIM 8
#define NEDGE 524288
#define HIDW 20

typedef __attribute__((ext_vector_type(16))) _Float16 v16h;
typedef __attribute__((ext_vector_type(8)))  _Float16 v8h;
typedef __attribute__((ext_vector_type(2)))  _Float16 h2v;
typedef __attribute__((ext_vector_type(8)))  float    v8f;

// A fragment: 16x32 f16, row-major tile in LDS, row stride ldk halves.
// Lane l: m = l&15; halves 0..7 = K=(l>>4)*8+0..7, halves 8..15 = K=16+(l>>4)*8+0..7.
__device__ __forceinline__ v16h wmma_load_a(const _Float16* tile, int ldk) {
  const int lane = threadIdx.x & 31;
  const int m = lane & 15;
  const int kg = (lane >> 4) << 3;
  const _Float16* row = tile + m * ldk;
  v8h lo = *(const v8h*)(row + kg);
  v8h hi = *(const v8h*)(row + 16 + kg);
  v16h a;
#pragma unroll
  for (int i = 0; i < 8; ++i) { a[i] = lo[i]; a[i + 8] = hi[i]; }
  return a;
}

// B fragment: 32x16 f16, stored N-major in LDS: tileT[n*ldk + k].
// Lane l: n = l&15; halves 0..7 = K=(l>>4)*8+0..7, halves 8..15 = K=16+(l>>4)*8+0..7.
__device__ __forceinline__ v16h wmma_load_b(const _Float16* tileT, int ldk) {
  const int lane = threadIdx.x & 31;
  const int n = lane & 15;
  const int k0 = (lane >> 4) << 3;
  const _Float16* col = tileT + n * ldk + k0;
  v8h lo = *(const v8h*)(col);
  v8h hi = *(const v8h*)(col + 16);
  v16h b;
#pragma unroll
  for (int i = 0; i < 8; ++i) { b[i] = lo[i]; b[i + 8] = hi[i]; }
  return b;
}

__device__ __forceinline__ v8f wmma16(v16h a, v16h b, v8f c) {
  return __builtin_amdgcn_wmma_f32_16x16x32_f16(false, a, false, b, (short)0, c,
                                                false, false);
}

__device__ __forceinline__ float sigmoidf_(float x) {
  return 1.0f / (1.0f + __expf(-x));
}

// ---------------------------------------------------------------------------
// Edge MLP + mean scatter.  128 threads = 4 waves, 16 edges per wave.
// phi: 136->20->20->128 padded to 160->32->32->128, all matmuls via WMMA f16.
// ---------------------------------------------------------------------------
__global__ __launch_bounds__(128) void edge_mlp_kernel(
    int mode, const float* __restrict__ x, const float* __restrict__ states,
    const int* __restrict__ src, const int* __restrict__ dst,
    const float* __restrict__ w1, const float* __restrict__ b1,
    const float* __restrict__ w2, const float* __restrict__ b2,
    const float* __restrict__ w3, const float* __restrict__ b3,
    float* __restrict__ aggr, float* __restrict__ cnt) {
  __shared__ __align__(16) _Float16 sW1[32 * 160];   // N-major [n][k]
  __shared__ __align__(16) _Float16 sW2[32 * 32];
  __shared__ __align__(16) _Float16 sW3[128 * 32];
  __shared__ float sB1[32], sB2[32], sB3[128];
  __shared__ __align__(16) _Float16 sA[4][16 * 160]; // per-wave A tiles
  __shared__ __align__(16) _Float16 sH1[4][16 * 32];
  __shared__ __align__(16) _Float16 sH2[4][16 * 32];
  __shared__ int sSrc[4][16], sDst[4][16];

  const int tid = threadIdx.x;
  // Stage weights transposed (N-major) with zero padding.
  for (int i = tid; i < 32 * 160; i += 128) {
    int n = i / 160, k = i - n * 160;
    sW1[i] = (k < 136 && n < HIDW) ? (_Float16)w1[k * HIDW + n] : (_Float16)0.f;
  }
  for (int i = tid; i < 32 * 32; i += 128) {
    int n = i >> 5, k = i & 31;
    sW2[i] = (k < HIDW && n < HIDW) ? (_Float16)w2[k * HIDW + n] : (_Float16)0.f;
  }
  for (int i = tid; i < 128 * 32; i += 128) {
    int n = i >> 5, k = i & 31;
    sW3[i] = (k < HIDW) ? (_Float16)w3[k * 128 + n] : (_Float16)0.f;
  }
  if (tid < 32) {
    sB1[tid] = (tid < HIDW) ? b1[tid] : 0.f;
    sB2[tid] = (tid < HIDW) ? b2[tid] : 0.f;
  }
  if (tid < 128) sB3[tid] = b3[tid];

  const int w = tid >> 5, lane = tid & 31;
  const int ebase = blockIdx.x * 64 + w * 16;
  if (lane < 16) {
    sSrc[w][lane] = src[ebase + lane];
    sDst[w][lane] = dst[ebase + lane];
  }
  __syncthreads();

  // Stage A: 16 edges x 160 (x[src] | states[src]-states[dst] | 0 pad), f16.
  for (int r = 0; r < 16; ++r) {
    const int s = sSrc[w][r], d = sDst[w][r];
    for (int k = lane; k < 160; k += 32) {
      float v;
      if (k < 128) {
        v = (mode == 0) ? 1.0f : fmaxf(x[s * FDIM + k], 0.f);
      } else if (k < 136) {
        v = states[s * SDIM + (k - 128)] - states[d * SDIM + (k - 128)];
      } else {
        v = 0.f;
      }
      sA[w][r * 160 + k] = (_Float16)v;
    }
  }
  __syncthreads();

  // Layer 1: 16x160 @ 160x32 (5 K-tiles x 2 N-tiles).
#pragma unroll
  for (int nt = 0; nt < 2; ++nt) {
    v8f acc = {};
#pragma unroll
    for (int kt = 0; kt < 5; ++kt) {
      v16h a = wmma_load_a(&sA[w][kt * 32], 160);
      v16h b = wmma_load_b(&sW1[(nt * 16) * 160 + kt * 32], 160);
      acc = wmma16(a, b, acc);
    }
    const int n = nt * 16 + (lane & 15);
#pragma unroll
    for (int r = 0; r < 8; ++r) {
      const int m = r + ((lane >> 4) << 3);
      sH1[w][m * 32 + n] = (_Float16)fmaxf(acc[r] + sB1[n], 0.f);
    }
  }
  __syncthreads();

  // Layer 2: 16x32 @ 32x32.
  {
    v16h a = wmma_load_a(&sH1[w][0], 32);
#pragma unroll
    for (int nt = 0; nt < 2; ++nt) {
      v8f acc = {};
      v16h b = wmma_load_b(&sW2[(nt * 16) * 32], 32);
      acc = wmma16(a, b, acc);
      const int n = nt * 16 + (lane & 15);
#pragma unroll
      for (int r = 0; r < 8; ++r) {
        const int m = r + ((lane >> 4) << 3);
        sH2[w][m * 32 + n] = (_Float16)fmaxf(acc[r] + sB2[n], 0.f);
      }
    }
  }
  __syncthreads();

  // Layer 3: 16x32 @ 32x128, scatter-add mean numerator to aggr[dst].
  {
    v16h a = wmma_load_a(&sH2[w][0], 32);
#pragma unroll
    for (int nt = 0; nt < 8; ++nt) {
      v8f acc = {};
      v16h b = wmma_load_b(&sW3[(nt * 16) * 32], 32);
      acc = wmma16(a, b, acc);
      const int n = nt * 16 + (lane & 15);
#pragma unroll
      for (int r = 0; r < 8; ++r) {
        const int m = r + ((lane >> 4) << 3);
        atomicAdd(&aggr[sDst[w][m] * PDIM + n], acc[r] + sB3[n]);
      }
    }
  }
  if (lane < 16) atomicAdd(&cnt[sDst[w][lane]], 1.0f);
}

// ---------------------------------------------------------------------------
__global__ void zero_kernel(float* __restrict__ p, int n) {
  int i = blockIdx.x * 256 + threadIdx.x;
  if (i < n) p[i] = 0.f;
}

__global__ void cvt_f16_kernel(_Float16* __restrict__ dst,
                               const float* __restrict__ src, int n) {
  int i = blockIdx.x * 256 + threadIdx.x;
  if (i < n) dst[i] = (_Float16)src[i];
}

// catbuf[n] = [ aggr[n]/max(cnt,1) | x part ]  as f16.
__global__ void concat_kernel(int mode, const float* __restrict__ aggr,
                              const float* __restrict__ cnt,
                              const float* __restrict__ x,
                              _Float16* __restrict__ catb) {
  int i = blockIdx.x * 256 + threadIdx.x;
  if (i >= N_AGENTS * 256) return;
  int node = i >> 8, c = i & 255;
  float v;
  if (c < PDIM) {
    v = aggr[node * PDIM + c] / fmaxf(cnt[node], 1.0f);
  } else {
    v = (mode == 0) ? 1.0f : fmaxf(x[node * FDIM + (c - PDIM)], 0.f);
  }
  catb[i] = (_Float16)v;
}

// ---------------------------------------------------------------------------
// Pre-gate GEMM: pre[4096,512] = catbuf[4096,256] @ wih^T + (bih+bhh).
// Block = 256 thr (8 waves), tile M=64, N=128, K staged 32 at a time.
// ---------------------------------------------------------------------------
__global__ __launch_bounds__(256) void pregate_gemm_kernel(
    const _Float16* __restrict__ A, const float* __restrict__ wih,
    const float* __restrict__ bih, const float* __restrict__ bhh,
    float* __restrict__ pre) {
  __shared__ __align__(16) _Float16 sAs[64 * 32];
  __shared__ __align__(16) _Float16 sBs[128 * 32]; // N-major [n][k]
  const int tid = threadIdx.x;
  const int M0 = blockIdx.x * 64, Nb = blockIdx.y * 128;
  const int w = tid >> 5, lane = tid & 31;
  const int mt = w & 3, nh = (w >> 2) * 64;
  v8f acc[4] = {{}, {}, {}, {}};
  for (int kc = 0; kc < 8; ++kc) {
    const int K0 = kc * 32;
    {  // stage A 64x32 (coalesced 16B vector copies)
      int r = tid >> 2, q = (tid & 3) * 8;
      *(v8h*)&sAs[r * 32 + q] = *(const v8h*)&A[(M0 + r) * 256 + K0 + q];
    }
    {  // stage B = wih^T chunk: Bs[n][k] = wih[Nb+n][K0+k], f32->f16
      int n = tid >> 1, hh = (tid & 1) * 16;
      const float* srcp = &wih[(Nb + n) * 256 + K0 + hh];
#pragma unroll
      for (int i = 0; i < 16; ++i) sBs[n * 32 + hh + i] = (_Float16)srcp[i];
    }
    __syncthreads();
    v16h a = wmma_load_a(&sAs[mt * 16 * 32], 32);
#pragma unroll
    for (int j = 0; j < 4; ++j) {
      v16h b = wmma_load_b(&sBs[(nh + j * 16) * 32], 32);
      acc[j] = wmma16(a, b, acc[j]);
    }
    __syncthreads();
  }
#pragma unroll
  for (int j = 0; j < 4; ++j) {
    const int gn = Nb + nh + j * 16 + (lane & 15);
    const float bsum = bih[gn] + bhh[gn];
#pragma unroll
    for (int r = 0; r < 8; ++r) {
      const int gm = M0 + mt * 16 + r + ((lane >> 4) << 3);
      pre[gm * 512 + gn] = acc[j][r] + bsum;
    }
  }
}

// ---------------------------------------------------------------------------
// Sequential 2-layer LSTM scan over 4096 steps, one workgroup of 512 threads.
// Thread j keeps Whh1/Wih2/Whh2 row j as packed-f16 register arrays (3x64
// VGPRs) so each step is pure VALU FMA against LDS-resident h vectors.
// pre already contains wih1@inp + bih1 + bhh1.
// ---------------------------------------------------------------------------
__global__ __launch_bounds__(512) void lstm_seq_kernel(
    const float* __restrict__ pre, const _Float16* __restrict__ whh1,
    const _Float16* __restrict__ wih2, const _Float16* __restrict__ whh2,
    const float* __restrict__ bih2, const float* __restrict__ bhh2,
    float* __restrict__ xout) {
  __shared__ float hs1[128], cs1[128], hs2[128], cs2[128];
  __shared__ float g1[512], g2[512];
  const int j = threadIdx.x;
  h2v wa[64], wb[64], wc[64];
  const h2v* pa = (const h2v*)(whh1 + j * 128);
  const h2v* pb = (const h2v*)(wih2 + j * 128);
  const h2v* pc = (const h2v*)(whh2 + j * 128);
#pragma unroll
  for (int i = 0; i < 64; ++i) { wa[i] = pa[i]; wb[i] = pb[i]; wc[i] = pc[i]; }
  const float b2s = bih2[j] + bhh2[j];
  if (j < 128) { hs1[j] = 0.f; cs1[j] = 0.f; hs2[j] = 0.f; cs2[j] = 0.f; }
  __syncthreads();

  for (int t = 0; t < N_AGENTS; ++t) {
    float g = pre[t * 512 + j];
    if (t + 1 < N_AGENTS) __builtin_prefetch(&pre[(t + 1) * 512 + j], 0, 0);
#pragma unroll
    for (int i = 0; i < 64; ++i)
      g += (float)wa[i].x * hs1[2 * i] + (float)wa[i].y * hs1[2 * i + 1];
    g1[j] = g;
    __syncthreads();
    if (j < 128) {  // torch gate order i,f,g,o
      float c = sigmoidf_(g1[128 + j]) * cs1[j] +
                sigmoidf_(g1[j]) * tanhf(g1[256 + j]);
      cs1[j] = c;
      hs1[j] = sigmoidf_(g1[384 + j]) * tanhf(c);
    }
    __syncthreads();
    float h = b2s;
#pragma unroll
    for (int i = 0; i < 64; ++i) {
      h += (float)wb[i].x * hs1[2 * i] + (float)wb[i].y * hs1[2 * i + 1];
      h += (float)wc[i].x * hs2[2 * i] + (float)wc[i].y * hs2[2 * i + 1];
    }
    g2[j] = h;
    __syncthreads();
    if (j < 128) {
      float c = sigmoidf_(g2[128 + j]) * cs2[j] +
                sigmoidf_(g2[j]) * tanhf(g2[256 + j]);
      cs2[j] = c;
      float hn = sigmoidf_(g2[384 + j]) * tanhf(c);
      hs2[j] = hn;
      xout[t * FDIM + j] = hn;
    }
    __syncthreads();
  }
}

// ---------------------------------------------------------------------------
// out[n] = relu(x[n]) . wout + bout   (one wave per node)
// ---------------------------------------------------------------------------
__global__ __launch_bounds__(256) void out_kernel(const float* __restrict__ x,
                                                  const float* __restrict__ wout,
                                                  const float* __restrict__ bout,
                                                  float* __restrict__ out) {
  const int node = (blockIdx.x * 256 + threadIdx.x) >> 5;
  const int lane = threadIdx.x & 31;
  if (node >= N_AGENTS) return;
  const float4 xv = *(const float4*)&x[node * FDIM + lane * 4];
  const float4 wv = *(const float4*)&wout[lane * 4];
  float s = fmaxf(xv.x, 0.f) * wv.x + fmaxf(xv.y, 0.f) * wv.y +
            fmaxf(xv.z, 0.f) * wv.z + fmaxf(xv.w, 0.f) * wv.w;
#pragma unroll
  for (int o = 16; o > 0; o >>= 1) s += __shfl_down(s, o, 32);
  if (lane == 0) out[node] = s + bout[0];
}

// ---------------------------------------------------------------------------
extern "C" void kernel_launch(void* const* d_in, const int* in_sizes, int n_in,
                              void* d_out, int out_size, void* d_ws,
                              size_t ws_size, hipStream_t stream) {
  (void)in_sizes; (void)n_in; (void)out_size; (void)ws_size;
  const float* states = (const float*)d_in[0];
  const int* eidx = (const int*)d_in[1];
  const int* src = eidx;
  const int* dst = eidx + NEDGE;

  // Workspace carve-up (~15 MB).
  float* aggr = (float*)d_ws;                      // [N,128]
  float* cnt = aggr + N_AGENTS * PDIM;             // [N]
  float* pre = cnt + N_AGENTS;                     // [N,512]
  float* xbuf = pre + N_AGENTS * 512;              // [N,128] layer output (raw h2)
  _Float16* catb = (_Float16*)(xbuf + N_AGENTS * FDIM);  // [N,256] f16
  _Float16* whh1h = catb + N_AGENTS * 256;               // [512,128] f16
  _Float16* wih2h = whh1h + 512 * 128;
  _Float16* whh2h = wih2h + 512 * 128;

  for (int L = 0; L < 2; ++L) {
    const int base = 2 + L * 14;
    const float* wih0 = (const float*)d_in[base + 0];
    const float* whh0 = (const float*)d_in[base + 1];
    const float* bih0 = (const float*)d_in[base + 2];
    const float* bhh0 = (const float*)d_in[base + 3];
    const float* wih1 = (const float*)d_in[base + 4];
    const float* whh1 = (const float*)d_in[base + 5];
    const float* bih1 = (const float*)d_in[base + 6];
    const float* bhh1 = (const float*)d_in[base + 7];
    const float* w1 = (const float*)d_in[base + 8];
    const float* b1 = (const float*)d_in[base + 9];
    const float* w2 = (const float*)d_in[base + 10];
    const float* b2 = (const float*)d_in[base + 11];
    const float* w3 = (const float*)d_in[base + 12];
    const float* b3 = (const float*)d_in[base + 13];
    const int mode = L;  // 0: x == ones, 1: x = relu(xbuf)

    zero_kernel<<<(N_AGENTS * (PDIM + 1) + 255) / 256, 256, 0, stream>>>(
        aggr, N_AGENTS * PDIM + N_AGENTS);
    edge_mlp_kernel<<<NEDGE / 64, 128, 0, stream>>>(
        mode, xbuf, states, src, dst, w1, b1, w2, b2, w3, b3, aggr, cnt);
    concat_kernel<<<(N_AGENTS * 256 + 255) / 256, 256, 0, stream>>>(
        mode, aggr, cnt, xbuf, catb);
    cvt_f16_kernel<<<(512 * 128 + 255) / 256, 256, 0, stream>>>(whh1h, whh0,
                                                                512 * 128);
    cvt_f16_kernel<<<(512 * 128 + 255) / 256, 256, 0, stream>>>(wih2h, wih1,
                                                                512 * 128);
    cvt_f16_kernel<<<(512 * 128 + 255) / 256, 256, 0, stream>>>(whh2h, whh1,
                                                                512 * 128);
    pregate_gemm_kernel<<<dim3(N_AGENTS / 64, 4), 256, 0, stream>>>(
        catb, wih0, bih0, bhh0, pre);
    lstm_seq_kernel<<<1, 512, 0, stream>>>(pre, whh1h, wih2h, whh2h, bih1,
                                           bhh1, xbuf);
  }

  const float* wout = (const float*)d_in[30];
  const float* bout = (const float*)d_in[31];
  out_kernel<<<N_AGENTS / 8, 256, 0, stream>>>(xbuf, wout, bout,
                                               (float*)d_out);
}
